// EAPNNForce_74096775791142
// MI455X (gfx1250) — matrix operand order repaired and checked by reading, not compiled, not run on hardware
//
#include <hip/hip_runtime.h>
#include <hip/hip_bf16.h>

// ---------------------------------------------------------------------------
// EAPNN forward for MI455X (gfx1250, wave32, WMMA).
// Pipeline: box_inv -> weight pack (f16, B-fragment order) -> per-atom ACSF ->
// per-pair APSF/features (f16, padded K=352) -> WMMA MLP (16 pairs / wave) ->
// deterministic block-sum reduction.
// Workspace use (~166 MB): binv, packed weights, G_atom (24MB),
// feats f16 (141MB), scale (0.8MB), block partial sums.
// ---------------------------------------------------------------------------

#define PI_F 3.14159265358979323846f

typedef __attribute__((ext_vector_type(8)))  _Float16 v8h;
typedef __attribute__((ext_vector_type(16))) _Float16 v16h;
typedef __attribute__((ext_vector_type(8)))  float    v8f;

#define WMMA_F16(A, B, C) \
  __builtin_amdgcn_wmma_f32_16x16x32_f16(false, (A), false, (B), (short)0, (C), false, false)

__constant__ float ZINDEX_d[10] = {1.f, 3.f, 5.f, 6.f, 7.f, 8.f, 9.f, 11.f, 15.f, 16.f};

// --------------------------- small helpers --------------------------------

__device__ __forceinline__ void pbc3(float dx, float dy, float dz,
                                     const float* __restrict__ box,
                                     const float* __restrict__ binv,
                                     float& ox, float& oy, float& oz) {
  float s0 = dx * binv[0] + dy * binv[3] + dz * binv[6];
  float s1 = dx * binv[1] + dy * binv[4] + dz * binv[7];
  float s2 = dx * binv[2] + dy * binv[5] + dz * binv[8];
  s0 -= floorf(s0 + 0.5f);
  s1 -= floorf(s1 + 0.5f);
  s2 -= floorf(s2 + 0.5f);
  ox = s0 * box[0] + s1 * box[3] + s2 * box[6];
  oy = s0 * box[1] + s1 * box[4] + s2 * box[7];
  oz = s0 * box[2] + s1 * box[5] + s2 * box[8];
}

// Branch-free scatter-add into 10 named accumulators (stays in VGPRs).
#define DECL10 float a0=0.f,a1=0.f,a2=0.f,a3=0.f,a4=0.f,a5=0.f,a6=0.f,a7=0.f,a8=0.f,a9=0.f
#define SEL10(t, e) do { float _e = (e); int _t = (t); \
  a0 += (_t==0)?_e:0.f; a1 += (_t==1)?_e:0.f; a2 += (_t==2)?_e:0.f; a3 += (_t==3)?_e:0.f; \
  a4 += (_t==4)?_e:0.f; a5 += (_t==5)?_e:0.f; a6 += (_t==6)?_e:0.f; a7 += (_t==7)?_e:0.f; \
  a8 += (_t==8)?_e:0.f; a9 += (_t==9)?_e:0.f; } while (0)

__device__ __forceinline__ v16h cat8(v8h lo, v8h hi) {
  v16h a;
#pragma unroll
  for (int i = 0; i < 8; ++i) { a[i] = lo[i]; a[i + 8] = hi[i]; }
  return a;
}

// B fragment: 16 contiguous halves per lane (two aligned b128 loads).
__device__ __forceinline__ v16h ldb(const _Float16* __restrict__ p) {
  return cat8(*reinterpret_cast<const v8h*>(p), *reinterpret_cast<const v8h*>(p + 8));
}

// ------------------------------ kernels ------------------------------------

__global__ void inv3_kernel(const float* __restrict__ b, float* __restrict__ o) {
  if (blockIdx.x == 0 && threadIdx.x == 0) {
    float a = b[0], B = b[1], c = b[2], d = b[3], e = b[4], f = b[5], g = b[6], h = b[7], i = b[8];
    float A0 = e * i - f * h, A1 = -(d * i - f * g), A2 = d * h - e * g;
    float det = a * A0 + B * A1 + c * A2;
    float id = 1.f / det;
    o[0] = A0 * id;            o[1] = -(B * i - c * h) * id; o[2] = (B * f - c * e) * id;
    o[3] = A1 * id;            o[4] = (a * i - c * g) * id;  o[5] = -(a * f - c * d) * id;
    o[6] = A2 * id;            o[7] = -(a * h - B * g) * id; o[8] = (a * e - B * d) * id;
  }
}

// Pack W (K x 64, f32) into fragment order:
// out[(((c*2+h)*4 + t)*16 + n)*16 + j] = (f16) W[(32c + 16h + j)*64 + 16t + n], zero-padded.
__global__ void pack_kernel(const float* __restrict__ W, _Float16* __restrict__ out,
                            int K, int chunks) {
  int idx = blockIdx.x * blockDim.x + threadIdx.x;
  int total = chunks * 2048;
  if (idx >= total) return;
  int j = idx & 15;
  int n = (idx >> 4) & 15;
  int t = (idx >> 8) & 3;
  int h = (idx >> 10) & 1;
  int c = idx >> 11;
  int k = 32 * c + 16 * h + j;
  int col = 16 * t + n;
  out[idx] = (k < K) ? (_Float16)W[(size_t)k * 64 + col] : (_Float16)0.f;
}

// Per-(atom, mu) radial ACSF: G_atom[n][k][type] (20x10 per atom).
__global__ __launch_bounds__(256) void atom_feat_kernel(
    const float* __restrict__ pos, const int* __restrict__ nbl,
    const int* __restrict__ atype, const float* __restrict__ box,
    const float* __restrict__ binv, float* __restrict__ G, int NA) {
  int idx = blockIdx.x * blockDim.x + threadIdx.x;
  if (idx >= NA * 20) return;
  int n = idx / 20, k = idx - n * 20;
  float mu = (5.0f / 19.0f) * (float)k;  // linspace(0, 5, 20)
  float px = pos[3 * n], py = pos[3 * n + 1], pz = pos[3 * n + 2];
  DECL10;
#pragma unroll
  for (int m = 0; m < 6; ++m) {
    int nb = nbl[n * 6 + m];
    if (nb >= 0) {
      float dx, dy, dz;
      pbc3(pos[3 * nb] - px, pos[3 * nb + 1] - py, pos[3 * nb + 2] - pz, box, binv, dx, dy, dz);
      float ax = dx + 1e-10f, ay = dy + 1e-10f, az = dz + 1e-10f;
      float dn = sqrtf(ax * ax + ay * ay + az * az);
      float fx = dn * 0.2f;
      float fc = (fx < 1.f) ? 0.5f * (__cosf(PI_F * fx) + 1.f) : 0.f;
      float dd = dn - mu;
      SEL10(atype[nb], __expf(-100.f * dd * dd) * fc);
    }
  }
  float* go = G + (size_t)n * 200 + (size_t)k * 10;
  go[0]=a0; go[1]=a1; go[2]=a2; go[3]=a3; go[4]=a4;
  go[5]=a5; go[6]=a6; go[7]=a7; go[8]=a8; go[9]=a9;
}

// Per-pair: cutoff scale + full 352-wide f16 feature row (322 real + 30 pad).
__global__ __launch_bounds__(128) void pair_feat_kernel(
    const float* __restrict__ pos, const float* __restrict__ box,
    const float* __restrict__ binv, const float* __restrict__ valid,
    const int* __restrict__ pairs, const int* __restrict__ nbl,
    const int* __restrict__ mol, const int* __restrict__ atype,
    const float* __restrict__ Gatom, _Float16* __restrict__ feats,
    float* __restrict__ scl, int NP) {
  int p = blockIdx.x * blockDim.x + threadIdx.x;
  if (p >= NP) return;

  int p0 = pairs[3 * p], p1 = pairs[3 * p + 1];
  int dp = (p1 - p0 <= 0) ? 1 : 0;
  p0 -= dp; p1 -= 2 * dp;
  float bs = (p0 < p1) ? valid[p] : 0.f;

  float rix = pos[3 * p0], riy = pos[3 * p0 + 1], riz = pos[3 * p0 + 2];
  float rjx = pos[3 * p1], rjy = pos[3 * p1 + 1], rjz = pos[3 * p1 + 2];
  float dx, dy, dz;
  pbc3(rjx - rix, rjy - riy, rjz - riz, box, binv, dx, dy, dz);
  float nx = dx + 1e-10f, ny = dy + 1e-10f, nz = dz + 1e-10f;
  float dr = sqrtf(nx * nx + ny * ny + nz * nz);
  float inter = (mol[p0] == mol[p1]) ? 0.f : 1.f;
  float cut = (dr <= 5.0f) ? 0.5f * (1.f + __cosf(PI_F * dr * 0.2f)) : 0.f;
  float scale = inter * bs * cut;
  scl[p] = scale;

  float uinv = 1.f / (dr + 1e-10f);
  float ux = dx * uinv, uy = dy * uinv, uz = dz * uinv;

  float cg[12], mk[12];
  int tp[12];
#pragma unroll
  for (int s = 0; s < 2; ++s) {
    int pc = s ? p1 : p0;
    int po = s ? p0 : p1;
    float cx = s ? rjx : rix, cy = s ? rjy : riy, cz = s ? rjz : riz;
    float sx = s ? -ux : ux,  sy = s ? -uy : uy,  sz = s ? -uz : uz;
#pragma unroll
    for (int m = 0; m < 6; ++m) {
      int nb = nbl[pc * 6 + m];
      bool ok = (nb >= 0) && (nb != pc) && (nb != po);
      float ex = 0.f, ey = 0.f, ez = 0.f;
      if (nb >= 0 && ok) { ex = pos[3 * nb]; ey = pos[3 * nb + 1]; ez = pos[3 * nb + 2]; }
      float rxx, rxy, rxz;
      pbc3(ex - cx, ey - cy, ez - cz, box, binv, rxx, rxy, rxz);
      float ax = rxx + 1e-10f, ay = rxy + 1e-10f, az = rxz + 1e-10f;
      float rn = sqrtf(ax * ax + ay * ay + az * az);
      float rinv = 1.f / rn;
      float c = (rxx * sx + rxy * sy + rxz * sz) * rinv;
      int q = s * 6 + m;
      cg[q] = ok ? c : 0.f;
      mk[q] = ok ? 1.f : 0.f;
      tp[q] = (nb >= 0) ? atype[nb] : 0;
    }
  }

  size_t base = (size_t)p * 352;

  // APSF pair block: feats[200 + k*10 + l]
  float fmul = 0.5f * scale;
#pragma unroll
  for (int k = 0; k < 10; ++k) {
    float mu = -1.f + (2.f / 9.f) * (float)k;  // linspace(-1, 1, 10)
    DECL10;
#pragma unroll
    for (int q = 0; q < 12; ++q) {
      float d = cg[q] - mu;
      SEL10(tp[q], __expf(-25.f * d * d) * mk[q]);
    }
    _Float16* fo = feats + base + 200 + k * 10;
    fo[0]=(_Float16)(a0*fmul); fo[1]=(_Float16)(a1*fmul); fo[2]=(_Float16)(a2*fmul);
    fo[3]=(_Float16)(a3*fmul); fo[4]=(_Float16)(a4*fmul); fo[5]=(_Float16)(a5*fmul);
    fo[6]=(_Float16)(a6*fmul); fo[7]=(_Float16)(a7*fmul); fo[8]=(_Float16)(a8*fmul);
    fo[9]=(_Float16)(a9*fmul);
  }

  // Atom feature block: 0.5*(G_atom[p0] + G_atom[p1])
  const float* ga = Gatom + (size_t)p0 * 200;
  const float* gb = Gatom + (size_t)p1 * 200;
  for (int f = 0; f < 200; ++f)
    feats[base + f] = (_Float16)(0.5f * (ga[f] + gb[f]));

  // Element block: [j_z, onehot_i(10), k_z, onehot_j(10)]
  int ti = atype[p0], tj = atype[p1];
  feats[base + 300] = (_Float16)ZINDEX_d[ti];
#pragma unroll
  for (int l = 0; l < 10; ++l) feats[base + 301 + l] = (_Float16)((ti == l) ? 1.f : 0.f);
  feats[base + 311] = (_Float16)ZINDEX_d[tj];
#pragma unroll
  for (int l = 0; l < 10; ++l) feats[base + 312 + l] = (_Float16)((tj == l) ? 1.f : 0.f);

  // K padding 322..351
#pragma unroll
  for (int f = 322; f < 352; ++f) feats[base + f] = (_Float16)0.f;
}

// LayerNorm + ReLU over one 16x64 tile (lanes 0..15 each own one row).
__device__ __forceinline__ void ln_relu_16(const float (*__restrict__ h)[68],
                                           _Float16 (*__restrict__ x)[72],
                                           const float* __restrict__ g,
                                           const float* __restrict__ be, int lane) {
  if (lane < 16) {
    const float* hr = h[lane];
    float s = 0.f;
#pragma unroll
    for (int j = 0; j < 64; ++j) s += hr[j];
    float mu = s * (1.f / 64.f);
    float v = 0.f;
#pragma unroll
    for (int j = 0; j < 64; ++j) { float d = hr[j] - mu; v += d * d; }
    float inv = rsqrtf(v * (1.f / 64.f) + 1e-6f);
#pragma unroll
    for (int j = 0; j < 64; ++j) {
      float y = (hr[j] - mu) * inv * g[j] + be[j];
      x[lane][j] = (_Float16)(y > 0.f ? y : 0.f);
    }
  }
}

// One 64->64 dense layer for a 16-pair tile: A from LDS (f16), B packed global.
__device__ __forceinline__ void dense64(const _Float16 (*__restrict__ xrow)[72],
                                        float (*__restrict__ h)[68],
                                        const _Float16* __restrict__ Wp,
                                        const float* __restrict__ b,
                                        int n16, int half) {
  v8f A0 = {}, A1 = {}, A2 = {}, A3 = {};
#pragma unroll
  for (int c = 0; c < 2; ++c) {
    const _Float16* xr0 = &xrow[n16][32 * c + half * 16];
    const _Float16* xr1 = &xrow[n16][32 * c + half * 16 + 8];
    v16h Af = cat8(*reinterpret_cast<const v8h*>(xr0), *reinterpret_cast<const v8h*>(xr1));
    const _Float16* wb = Wp + (size_t)(c * 2 + half) * 1024 + n16 * 16;
    A0 = WMMA_F16(Af, ldb(wb), A0);
    A1 = WMMA_F16(Af, ldb(wb + 256), A1);
    A2 = WMMA_F16(Af, ldb(wb + 512), A2);
    A3 = WMMA_F16(Af, ldb(wb + 768), A3);
  }
  float b0 = b[n16], b1v = b[16 + n16], b2v = b[32 + n16], b3v = b[48 + n16];
#pragma unroll
  for (int r = 0; r < 8; ++r) {
    int row = r + 8 * half;
    h[row][n16]      = A0[r] + b0;
    h[row][16 + n16] = A1[r] + b1v;
    h[row][32 + n16] = A2[r] + b2v;
    h[row][48 + n16] = A3[r] + b3v;
  }
}

// MLP: 4 wave32s per block, each wave owns a 16-pair tile.
__global__ __launch_bounds__(128) void mlp_kernel(
    const _Float16* __restrict__ feats, const float* __restrict__ scl,
    const _Float16* __restrict__ W1p, const _Float16* __restrict__ W2p,
    const _Float16* __restrict__ W3p,
    const float* __restrict__ b1, const float* __restrict__ g1, const float* __restrict__ be1,
    const float* __restrict__ b2, const float* __restrict__ g2, const float* __restrict__ be2,
    const float* __restrict__ b3, const float* __restrict__ g3, const float* __restrict__ be3,
    const float* __restrict__ Wo, const float* __restrict__ bo,
    float* __restrict__ bsums, int NP) {
  __shared__ __align__(16) float    hb[4][16][68];
  __shared__ __align__(16) _Float16 xb[4][16][72];
  __shared__ float ps[64];

  const int lane = threadIdx.x & 31;
  const int wave = threadIdx.x >> 5;
  const int n16 = lane & 15;
  const int half = lane >> 4;
  const long tile = (long)blockIdx.x * 4 + wave;
  long pidx = tile * 16 + n16;
  if (pidx > (long)NP - 1) pidx = (long)NP - 1;  // clamp reads for tail tiles

  // -------- layer 1: x(16x352 f16) @ W1(352x64) --------
  v8f c0 = {}, c1 = {}, c2 = {}, c3 = {};
  const _Float16* arow = feats + pidx * 352;
#pragma unroll
  for (int c = 0; c < 11; ++c) {
    v16h Af = cat8(*reinterpret_cast<const v8h*>(arow + 32 * c + half * 16),
                   *reinterpret_cast<const v8h*>(arow + 32 * c + half * 16 + 8));
    const _Float16* wb = W1p + (size_t)(c * 2 + half) * 1024 + n16 * 16;
    c0 = WMMA_F16(Af, ldb(wb),       c0);
    c1 = WMMA_F16(Af, ldb(wb + 256), c1);
    c2 = WMMA_F16(Af, ldb(wb + 512), c2);
    c3 = WMMA_F16(Af, ldb(wb + 768), c3);
  }
  {
    float bt0 = b1[n16], bt1 = b1[16 + n16], bt2 = b1[32 + n16], bt3 = b1[48 + n16];
#pragma unroll
    for (int r = 0; r < 8; ++r) {
      int row = r + 8 * half;
      hb[wave][row][n16]      = c0[r] + bt0;
      hb[wave][row][16 + n16] = c1[r] + bt1;
      hb[wave][row][32 + n16] = c2[r] + bt2;
      hb[wave][row][48 + n16] = c3[r] + bt3;
    }
  }
  __syncthreads();
  ln_relu_16(hb[wave], xb[wave], g1, be1, lane);
  __syncthreads();

  // -------- layers 2 and 3 --------
  dense64(xb[wave], hb[wave], W2p, b2, n16, half);
  __syncthreads();
  ln_relu_16(hb[wave], xb[wave], g2, be2, lane);
  __syncthreads();

  dense64(xb[wave], hb[wave], W3p, b3, n16, half);
  __syncthreads();
  ln_relu_16(hb[wave], xb[wave], g3, be3, lane);
  __syncthreads();

  // -------- output head + weighted sum --------
  if (lane < 16) {
    float d = 0.f;
#pragma unroll
    for (int j = 0; j < 64; ++j) d += (float)xb[wave][lane][j] * Wo[j];
    long pi = tile * 16 + lane;
    float sc = (pi < (long)NP) ? scl[pi] : 0.f;
    ps[wave * 16 + lane] = (d + bo[0]) * sc;
  }
  __syncthreads();
  if (threadIdx.x == 0) {
    float s = 0.f;
    for (int i = 0; i < 64; ++i) s += ps[i];
    bsums[blockIdx.x] = s;
  }
}

__global__ __launch_bounds__(256) void reduce_kernel(const float* __restrict__ in, int n,
                                                     float* __restrict__ out) {
  __shared__ float sm[256];
  int tid = threadIdx.x;
  float s = 0.f;
  for (int i = tid; i < n; i += 256) s += in[i];
  sm[tid] = s;
  __syncthreads();
  for (int st = 128; st > 0; st >>= 1) {
    if (tid < st) sm[tid] += sm[tid + st];
    __syncthreads();
  }
  if (tid == 0) out[0] = sm[0];
}

// ------------------------------ launcher -----------------------------------

extern "C" void kernel_launch(void* const* d_in, const int* in_sizes, int n_in,
                              void* d_out, int out_size, void* d_ws, size_t ws_size,
                              hipStream_t stream) {
  (void)n_in; (void)out_size; (void)ws_size;

  const float* pos   = (const float*)d_in[0];
  const float* box   = (const float*)d_in[1];
  const float* valid = (const float*)d_in[2];
  const float* W1    = (const float*)d_in[3];
  const float* b1    = (const float*)d_in[4];
  const float* g1    = (const float*)d_in[5];
  const float* be1   = (const float*)d_in[6];
  const float* W2    = (const float*)d_in[7];
  const float* b2    = (const float*)d_in[8];
  const float* g2    = (const float*)d_in[9];
  const float* be2   = (const float*)d_in[10];
  const float* W3    = (const float*)d_in[11];
  const float* b3    = (const float*)d_in[12];
  const float* g3    = (const float*)d_in[13];
  const float* be3   = (const float*)d_in[14];
  const float* Wo    = (const float*)d_in[15];
  const float* bo    = (const float*)d_in[16];
  const int* pairs   = (const int*)d_in[17];
  const int* nbl     = (const int*)d_in[18];
  // d_in[19] = topo_mask: redundant with (topo_nblist != -1), unused.
  const int* mol     = (const int*)d_in[20];
  const int* atype   = (const int*)d_in[21];

  const int NP = in_sizes[2];   // pairs
  const int NA = in_sizes[20];  // atoms

  // workspace carve-out (256B aligned blocks)
  char* ws = (char*)d_ws;
  size_t off = 0;
  auto carve = [&](size_t bytes) -> char* {
    char* p = ws + off;
    off = (off + bytes + 255) & ~(size_t)255;
    return p;
  };
  float*     binv  = (float*)carve(9 * sizeof(float));
  _Float16*  W1p   = (_Float16*)carve((size_t)11 * 2048 * sizeof(_Float16));
  _Float16*  W2p   = (_Float16*)carve((size_t)2 * 2048 * sizeof(_Float16));
  _Float16*  W3p   = (_Float16*)carve((size_t)2 * 2048 * sizeof(_Float16));
  float*     Gatom = (float*)carve((size_t)NA * 200 * sizeof(float));
  _Float16*  feats = (_Float16*)carve((size_t)NP * 352 * sizeof(_Float16));
  float*     scl   = (float*)carve((size_t)NP * sizeof(float));
  const int ntiles = (NP + 15) / 16;
  const int nblk   = (ntiles + 3) / 4;
  float*     bsums = (float*)carve((size_t)nblk * sizeof(float));

  inv3_kernel<<<1, 1, 0, stream>>>(box, binv);

  pack_kernel<<<(11 * 2048 + 255) / 256, 256, 0, stream>>>(W1, W1p, 322, 11);
  pack_kernel<<<(2 * 2048 + 255) / 256, 256, 0, stream>>>(W2, W2p, 64, 2);
  pack_kernel<<<(2 * 2048 + 255) / 256, 256, 0, stream>>>(W3, W3p, 64, 2);

  atom_feat_kernel<<<(NA * 20 + 255) / 256, 256, 0, stream>>>(pos, nbl, atype, box, binv,
                                                              Gatom, NA);

  pair_feat_kernel<<<(NP + 127) / 128, 128, 0, stream>>>(pos, box, binv, valid, pairs, nbl,
                                                         mol, atype, Gatom, feats, scl, NP);

  mlp_kernel<<<nblk, 128, 0, stream>>>(feats, scl, W1p, W2p, W3p,
                                       b1, g1, be1, b2, g2, be2, b3, g3, be3,
                                       Wo, bo, bsums, NP);

  reduce_kernel<<<1, 256, 0, stream>>>(bsums, nblk, (float*)d_out);
}
